// SchLoss_40407052321227
// MI455X (gfx1250) — compile-verified, matrix operand order checked by reading, not run on hardware
//
#include <hip/hip_runtime.h>
#include <hip/hip_bf16.h>
#include <math.h>

// Problem constants (from the reference)
#define B_    4
#define C_    16
#define H_    152
#define W_    272
#define M_    32
#define P_    (H_ * W_)        // 41344 pixels
#define WCH_  593
// 41344 = 19 tiles * 34 iters * 64 px ; each of 4 waves does 16 px / iter.
// Iterations processed in pairs so the loss epilogue uses all 32 lanes.
#define ITER_   34
#define PAIRS_  (ITER_ / 2)
#define TILES_  19
#define WGSIZE  128

typedef __attribute__((ext_vector_type(16))) __bf16 v16bf;
typedef __attribute__((ext_vector_type(8)))  float  v8f;

union Frag {
    unsigned int u[8];
    v16bf bf;
};

// Pack two f32 into packed bf16 (truncation) with a single v_perm_b32:
// result = { hi[31:16], lo[31:16] }
__device__ __forceinline__ unsigned int pack2(float lo, float hi) {
    return __builtin_amdgcn_perm(__float_as_uint(hi), __float_as_uint(lo),
                                 0x07060302u);
}
__device__ __forceinline__ unsigned short bf16h(float f) {
    return (unsigned short)(__float_as_uint(f) >> 16);
}
// relu in one v_med3_num_f32: med3(v, 0, +inf) clamps below at 0
__device__ __forceinline__ float relu1(float v) {
    return __builtin_amdgcn_fmed3f(v, 0.f, __builtin_inff());
}

#define BF16_ONE 0x00003f80u   // bf16(1.0) in the low half of a pair

// acc[0] = pos_loss sum, acc[1] = neg_loss sum
__global__ void __launch_bounds__(WGSIZE)
sch_loss_main(const float* __restrict__ sch_feat,
              const float* __restrict__ conv_weight,
              const float* __restrict__ mask,
              const int*   __restrict__ pre_ind,
              const float* __restrict__ target,
              const int*   __restrict__ ind,
              float*       __restrict__ acc) {
    __shared__ float wraw[WCH_];                             // per-object weights
    __shared__ __align__(16) unsigned short h16[4][16 * 16]; // [wave][px*16+ch] bf16
    __shared__ float zbuf[4][32];                            // 2 iters of logits
    __shared__ float red[4];                                 // per-wave partials

    const int wg   = blockIdx.x;
    const int obj  = wg / TILES_;         // b*M + m
    const int tile = wg % TILES_;
    const int b    = obj / M_;
    const int tid  = threadIdx.x;
    const int wave = tid >> 5;
    const int lane = tid & 31;
    const int ln16 = lane & 15;
    const bool hi  = (lane >= 16);

    const int pre = pre_ind[obj];

    // ---- cooperative gather of 593 weights (stride-P_ in conv_weight) ----
    const float* cwb = conv_weight + (size_t)b * WCH_ * P_ + pre;
    for (int j = tid; j < WCH_; j += WGSIZE) wraw[j] = cwb[(size_t)j * P_];
    __syncthreads();

    // ---- build bf16 B fragments in registers (one-time) ----
    // 16-bit B (32x16) layout (mirror of A): lane -> column n = lane%16;
    // lanes 0-15 hold K=0..7 (v0..3, packed pairs) and K=16..23 (v4..7);
    // lanes 16-31 hold K=8..15 and K=24..31.
    // Layer1 K map: 0..15 = feature channels, 16/17 = x_rel/y_rel coefs,
    // 18 = bias row (A supplies 1.0). Layer2: 0..15 = channels, 16 = bias.
    // Layer3: same, only column n==0 non-zero.
    Frag b1f, b2f, b3f;
    {
        const int n = ln16;
        const int ko = hi ? 8 : 0;                 // K offset for this lane half
        #pragma unroll
        for (int r = 0; r < 4; ++r)
            b1f.u[r] = pack2(wraw[n * 18 + ko + 2 * r],
                             wraw[n * 18 + ko + 2 * r + 1]);
        b1f.u[4] = hi ? 0u : pack2(wraw[n * 18 + 16], wraw[n * 18 + 17]);
        b1f.u[5] = hi ? 0u : pack2(wraw[288 + n], 0.f);
        b1f.u[6] = 0u; b1f.u[7] = 0u;

        #pragma unroll
        for (int r = 0; r < 4; ++r)
            b2f.u[r] = pack2(wraw[304 + n * 16 + ko + 2 * r],
                             wraw[304 + n * 16 + ko + 2 * r + 1]);
        b2f.u[4] = hi ? 0u : pack2(wraw[560 + n], 0.f);
        b2f.u[5] = 0u; b2f.u[6] = 0u; b2f.u[7] = 0u;

        #pragma unroll
        for (int r = 0; r < 4; ++r) {
            const unsigned int v = pack2(wraw[576 + ko + 2 * r],
                                         wraw[576 + ko + 2 * r + 1]);
            b3f.u[r] = (n == 0) ? v : 0u;
        }
        b3f.u[4] = (!hi && n == 0) ? pack2(wraw[592], 0.f) : 0u;
        b3f.u[5] = 0u; b3f.u[6] = 0u; b3f.u[7] = 0u;
    }

    const float xo = (float)(pre % W_);
    const float yo = (float)(pre / W_);
    const float* featb = sch_feat + (size_t)b * C_ * P_ + (hi ? (size_t)8 * P_ : 0);
    const float* tgtb  = target + (size_t)obj * P_;
    const int    indv  = ind[obj];
    const float  maskv = mask[obj];

    float negAcc = 0.f;
    const int p0 = tile * (ITER_ * 64) + wave * 16 + ln16;  // lane's pixel, iter 0
    int x = p0 % W_;
    int y = p0 / W_;
    unsigned short* hw = &h16[wave][0];
    const unsigned short* hrd = &h16[wave][ln16 * 16 + (hi ? 8 : 0)];
    const int sbase = (hi ? 8 : 0) * 16 + ln16;             // LDS halfword store base

    for (int j = 0; j < PAIRS_; ++j) {
        #pragma unroll
        for (int s = 0; s < 2; ++s) {
            const int p = p0 + (2 * j + s) * 64;

            __builtin_prefetch(featb + (p + 64), 0, 0);   // global_prefetch_b8
            __builtin_prefetch(tgtb + (p + 64), 0, 0);

            // ---- A1: 16 px x (16 ch + x_rel + y_rel + 1), branchless ----
            Frag a1;
            {
                const float* fp = featb + p;
                #pragma unroll
                for (int r = 0; r < 4; ++r)
                    a1.u[r] = pack2(fp[(size_t)(2 * r) * P_],
                                    fp[(size_t)(2 * r + 1) * P_]);
                const float xr = ((float)x - xo) * (1.f / 128.f);
                const float yr = ((float)y - yo) * (1.f / 128.f);
                a1.u[4] = hi ? 0u : pack2(xr, yr);   // K16,K17
                a1.u[5] = hi ? 0u : BF16_ONE;        // K18 -> bias row
                a1.u[6] = 0u; a1.u[7] = 0u;
            }
            // incremental x/y update (64 < W so one conditional subtract)
            x += 64;
            if (x >= W_) { x -= W_; ++y; }

            v8f c1 = {};
            c1 = __builtin_amdgcn_wmma_f32_16x16x32_bf16(
                     false, a1.bf, false, b1f.bf, (short)0, c1, false, false);

            // relu (v_med3) + bf16 high-half store (ds_store_b16_d16_hi)
            #pragma unroll
            for (int r = 0; r < 8; ++r)
                hw[sbase + r * 16] = bf16h(relu1(c1[r]));

            // ---- A2: one ds_load_b128 gives 4 pre-packed channel pairs ----
            Frag a2;
            {
                const uint4 q = *(const uint4*)hrd;
                a2.u[0] = q.x; a2.u[1] = q.y; a2.u[2] = q.z; a2.u[3] = q.w;
                a2.u[4] = hi ? 0u : BF16_ONE;  // K16 bias slot
                a2.u[5] = 0u; a2.u[6] = 0u; a2.u[7] = 0u;
            }

            v8f c2 = {};
            c2 = __builtin_amdgcn_wmma_f32_16x16x32_bf16(
                     false, a2.bf, false, b2f.bf, (short)0, c2, false, false);

            #pragma unroll
            for (int r = 0; r < 8; ++r)
                hw[sbase + r * 16] = bf16h(relu1(c2[r]));

            Frag a3;
            {
                const uint4 q = *(const uint4*)hrd;
                a3.u[0] = q.x; a3.u[1] = q.y; a3.u[2] = q.z; a3.u[3] = q.w;
                a3.u[4] = hi ? 0u : BF16_ONE;
                a3.u[5] = 0u; a3.u[6] = 0u; a3.u[7] = 0u;
            }

            v8f c3 = {};
            c3 = __builtin_amdgcn_wmma_f32_16x16x32_bf16(
                     false, a3.bf, false, b3f.bf, (short)0, c3, false, false);

            // logits in column n=0: lane 0 -> px 0..7, lane 16 -> px 8..15
            if (ln16 == 0) {
                const int mb = (hi ? 8 : 0) + s * 16;
                #pragma unroll
                for (int r = 0; r < 8; ++r) zbuf[wave][mb + r] = c3[r];
            }
        }

        // ---- batched loss: all 32 lanes, 32 pixels (two 64B target segs) ----
        {
            const int pz = p0 + (2 * j + (hi ? 1 : 0)) * 64;
            const float z   = zbuf[wave][lane];
            const float e   = __expf(-z);          // v_exp_f32
            const float hm  = __frcp_rn(1.f + e);  // sigmoid
            const float l1e = __logf(1.f + e);     // log(1+e)
            // log1p(-hm) = -z - log(1+e);  log(hm) = -log(1+e)
            const float t    = tgtb[pz];
            const float omt  = 1.f - t;
            const float omt2 = omt * omt;
            negAcc += (-z - l1e) * hm * hm * (omt2 * omt2);
            if (pz == indv) {
                const float d = 1.f - hm;
                atomicAdd(&acc[0], (-l1e) * d * d * maskv);
            }
        }
    }

    // wave32 shuffle reduction, then one global atomic per workgroup
    float s = negAcc;
    #pragma unroll
    for (int off = 16; off > 0; off >>= 1) s += __shfl_xor(s, off, 32);
    if (lane == 0) red[wave] = s;
    __syncthreads();
    if (tid == 0) atomicAdd(&acc[1], red[0] + red[1] + red[2] + red[3]);
}

__global__ void sch_loss_init(float* acc) {
    if (threadIdx.x < 2) acc[threadIdx.x] = 0.f;
}

__global__ void sch_loss_final(const float* __restrict__ mask,
                               const float* __restrict__ acc,
                               float* __restrict__ out) {
    float np = 0.f;
    for (int i = 0; i < B_ * M_; ++i) np += mask[i];
    const float pos = acc[0], neg = acc[1];
    out[0] = (np == 0.f) ? -neg : -(pos + neg) / np;
}

extern "C" void kernel_launch(void* const* d_in, const int* in_sizes, int n_in,
                              void* d_out, int out_size, void* d_ws, size_t ws_size,
                              hipStream_t stream) {
    const float* sch_feat    = (const float*)d_in[0];
    const float* conv_weight = (const float*)d_in[1];
    const float* mask        = (const float*)d_in[2];
    const int*   pre_ind     = (const int*)d_in[3];
    const float* target      = (const float*)d_in[4];
    const int*   ind         = (const int*)d_in[5];
    float* acc = (float*)d_ws;
    float* out = (float*)d_out;

    sch_loss_init<<<1, 32, 0, stream>>>(acc);
    sch_loss_main<<<dim3(B_ * M_ * TILES_), dim3(WGSIZE), 0, stream>>>(
        sch_feat, conv_weight, mask, pre_ind, target, ind, acc);
    sch_loss_final<<<1, 1, 0, stream>>>(mask, acc, out);
}
